// GCN_6605659702082
// MI455X (gfx1250) — matrix operand order repaired and checked by reading, not compiled, not run on hardware
//
#include <hip/hip_runtime.h>
#include <math.h>

#define N_FEAT 64
#define N_CLASSES 10
#define N_GRAPHS 64
#define BN_EPS 1e-5f
#define TILE_M 128
#define APITCH 72   // bf16 elements per LDS row (144 B, keeps 16B alignment)

typedef __attribute__((ext_vector_type(16))) __bf16 v16bf;
typedef __attribute__((ext_vector_type(8)))  __bf16 v8bf;    // 16 bytes
typedef __attribute__((ext_vector_type(8)))  float  v8f;

union Frag16 { v16bf v; v8bf h[2]; };

__device__ __forceinline__ unsigned short f32_to_bf16_bits(float f) {
    union { float f; unsigned u; } in; in.f = f;
    unsigned r = in.u + 0x7FFFu + ((in.u >> 16) & 1u);   // round to nearest even
    return (unsigned short)(r >> 16);
}

// inverse of the 16-bit A 16x32 fragment permutation (ISA 7.12.2):
// element position within a 64-wide permuted row for original column k
__device__ __forceinline__ int a_perm(int k) {
    return (k >> 5) * 32 + ((k >> 3) & 1) * 16 + ((k >> 4) & 1) * 8 + (k & 7);
}

// ---------------- degree / normalization ----------------
__global__ void deg_init_k(float* __restrict__ deg, int n) {
    int i = blockIdx.x * blockDim.x + threadIdx.x;
    if (i < n) deg[i] = 1.0f;                  // self-loop
}
__global__ void deg_accum_k(const int* __restrict__ dst, float* __restrict__ deg, int e) {
    int i = blockIdx.x * blockDim.x + threadIdx.x;
    if (i < e) atomicAdd(&deg[dst[i]], 1.0f);
}
__global__ void isq_k(const float* __restrict__ deg, float* __restrict__ isq, int n) {
    int i = blockIdx.x * blockDim.x + threadIdx.x;
    if (i < n) isq[i] = rsqrtf(fmaxf(deg[i], 1e-12f));
}
__global__ void norm_k(const int* __restrict__ src, const int* __restrict__ dst,
                       const float* __restrict__ isq, float* __restrict__ nrm, int e) {
    int i = blockIdx.x * blockDim.x + threadIdx.x;
    if (i < e) nrm[i] = isq[src[i]] * isq[dst[i]];
}

// ---------------- dense transform: OUT = H @ W (bf16 WMMA, f32 accum) ----------------
// block = 256 threads = 8 waves; each wave computes a 16(M) x 64(N) tile; K = 64.
// sA holds the H tile pre-permuted into A-fragment order; sBt holds W transposed,
// so every fragment is two contiguous 16-byte LDS loads. Row-bound guards are
// block-uniform: only the final partial tile takes the guarded path.
__global__ void gemm_wmma_k(const float* __restrict__ H, const float* __restrict__ W,
                            float* __restrict__ OUT, int n) {
    __shared__ __bf16 sA [TILE_M][APITCH];   // permuted H tile
    __shared__ __bf16 sBt[64][APITCH];       // W transposed: sBt[col][k]

    const int tid = threadIdx.x;
    const int rowBase = blockIdx.x * TILE_M;
    const bool fullTile = (rowBase + TILE_M) <= n;   // uniform across block

    // --- fill A: 4096 bf16-pairs; coalesced float2 global reads, packed u32 LDS stores
    if (fullTile) {
        #pragma unroll
        for (int i = 0; i < (TILE_M * 32) / 256; ++i) {   // 16 iters
            int p = tid + i * 256;
            int r = p >> 5;
            int c = (p & 31) * 2;
            float2 v = *(const float2*)(H + (rowBase + r) * 64 + c);
            unsigned lo = f32_to_bf16_bits(v.x);
            unsigned hiB = f32_to_bf16_bits(v.y);
            *(unsigned*)((char*)&sA[r][0] + a_perm(c) * 2) = lo | (hiB << 16);
        }
    } else {
        #pragma unroll
        for (int i = 0; i < (TILE_M * 32) / 256; ++i) {
            int p = tid + i * 256;
            int r = p >> 5;
            int c = (p & 31) * 2;
            int row = rowBase + r;
            float2 v = make_float2(0.0f, 0.0f);
            if (row < n) v = *(const float2*)(H + row * 64 + c);
            unsigned lo = f32_to_bf16_bits(v.x);
            unsigned hiB = f32_to_bf16_bits(v.y);
            *(unsigned*)((char*)&sA[r][0] + a_perm(c) * 2) = lo | (hiB << 16);
        }
    }
    // --- fill B transposed: 2048 pairs over k; store u32 at sBt[c][k], k even
    #pragma unroll
    for (int i = 0; i < 2048 / 256; ++i) {                // 8 iters
        int p = tid + i * 256;
        int c = p & 63;
        int k = (p >> 6) * 2;
        unsigned lo = f32_to_bf16_bits(W[k * 64 + c]);
        unsigned hiB = f32_to_bf16_bits(W[(k + 1) * 64 + c]);
        *(unsigned*)((char*)&sBt[c][0] + k * 2) = lo | (hiB << 16);
    }
    __syncthreads();

    const int w     = tid >> 5;        // wave id 0..7
    const int lane  = tid & 31;
    const int m     = lane & 15;       // fragment row / column index
    const int hi    = lane >> 4;       // lane half
    const int waveM = w * 16;

    v8f acc[4] = {v8f{}, v8f{}, v8f{}, v8f{}};

    #pragma unroll
    for (int ks = 0; ks < 2; ++ks) {
        // A fragment: contiguous 32 bytes at permuted offset (2x ds_load_b128)
        Frag16 fa;
        const __bf16* pa = &sA[waveM + m][ks * 32 + hi * 16];
        fa.h[0] = *(const v8bf*)(pa);
        fa.h[1] = *(const v8bf*)(pa + 8);
        #pragma unroll
        for (int nt = 0; nt < 4; ++nt) {
            // B fragment: column nt*16+m of W, K = ks*32 + hi*8 .. +7 and ks*32 + 16 + hi*8 .. +7
            Frag16 fb;
            const __bf16* pb = &sBt[nt * 16 + m][ks * 32 + hi * 8];
            fb.h[0] = *(const v8bf*)(pb);
            fb.h[1] = *(const v8bf*)(pb + 16);
            acc[nt] = __builtin_amdgcn_wmma_f32_16x16x32_bf16(
                false, fa.v, false, fb.v, (short)0, acc[nt], false, false);
        }
    }

    // C/D layout: VGPR r -> M = r + 8*hi, N = lane&15 (per 16-col tile).
    // Per (r,nt) a wave's stores cover two contiguous 64B runs -> coalesced b32s.
    if (fullTile) {
        #pragma unroll
        for (int nt = 0; nt < 4; ++nt)
            #pragma unroll
            for (int r = 0; r < 8; ++r)
                OUT[(rowBase + waveM + r + 8 * hi) * 64 + nt * 16 + m] = acc[nt][r];
    } else {
        #pragma unroll
        for (int nt = 0; nt < 4; ++nt)
            #pragma unroll
            for (int r = 0; r < 8; ++r) {
                int grow = rowBase + waveM + r + 8 * hi;
                if (grow < n) OUT[grow * 64 + nt * 16 + m] = acc[nt][r];
            }
    }
}

// ---------------- sparse aggregation ----------------
// agg[i,f] = hl[i,f]*isq[i]^2 + bias[f]   (self-loop term + bias folded in)
// one thread per (node, 4 feats): float4 loads/stores
__global__ void selfloop_bias_k(const float* __restrict__ hl, const float* __restrict__ isq,
                                const float* __restrict__ bias, float* __restrict__ agg, int n) {
    int idx = blockIdx.x * blockDim.x + threadIdx.x;
    if (idx < n * 16) {
        int i = idx >> 4, fg = (idx & 15) * 4;
        float s = isq[i];
        float s2 = s * s;
        float4 v = *(const float4*)(hl + i * 64 + fg);
        float4 b = *(const float4*)(bias + fg);
        float4 o = make_float4(v.x * s2 + b.x, v.y * s2 + b.y,
                               v.z * s2 + b.z, v.w * s2 + b.w);
        *(float4*)(agg + i * 64 + fg) = o;
    }
}
// agg[dst, f..f+3] += hl[src, f..f+3] * norm[e]   (b128 gather, 4 f32 atomics)
__global__ void scatter_k(const int* __restrict__ src, const int* __restrict__ dst,
                          const float* __restrict__ nrm, const float* __restrict__ hl,
                          float* __restrict__ agg, int total) {
    int idx = blockIdx.x * blockDim.x + threadIdx.x;
    if (idx < total) {
        int e = idx >> 4, fg = (idx & 15) * 4;
        float w = nrm[e];
        float4 v = *(const float4*)(hl + src[e] * 64 + fg);
        float* p = agg + dst[e] * 64 + fg;
        atomicAdd(p + 0, v.x * w);
        atomicAdd(p + 1, v.y * w);
        atomicAdd(p + 2, v.z * w);
        atomicAdd(p + 3, v.w * w);
    }
}

// ---------------- batch norm (train-mode, fresh stats) ----------------
__global__ void zero_k(float* __restrict__ p, int n) {
    int i = blockIdx.x * blockDim.x + threadIdx.x;
    if (i < n) p[i] = 0.0f;
}
__global__ void bn_stats_k(const float* __restrict__ agg, float* __restrict__ stats,
                           int n, int do_relu) {
    int t = blockIdx.x * blockDim.x + threadIdx.x;
    int fg = (t & 15) * 4;
    int row0 = t >> 4;
    int stride = (gridDim.x * blockDim.x) >> 4;
    float s0 = 0.f, s1 = 0.f, s2 = 0.f, s3 = 0.f;
    float q0 = 0.f, q1 = 0.f, q2 = 0.f, q3 = 0.f;
    for (int i = row0; i < n; i += stride) {
        float4 v = *(const float4*)(agg + i * 64 + fg);
        if (do_relu) {
            v.x = fmaxf(v.x, 0.f); v.y = fmaxf(v.y, 0.f);
            v.z = fmaxf(v.z, 0.f); v.w = fmaxf(v.w, 0.f);
        }
        s0 += v.x; s1 += v.y; s2 += v.z; s3 += v.w;
        q0 += v.x * v.x; q1 += v.y * v.y; q2 += v.z * v.z; q3 += v.w * v.w;
    }
    atomicAdd(&stats[fg + 0], s0); atomicAdd(&stats[fg + 1], s1);
    atomicAdd(&stats[fg + 2], s2); atomicAdd(&stats[fg + 3], s3);
    atomicAdd(&stats[64 + fg + 0], q0); atomicAdd(&stats[64 + fg + 1], q1);
    atomicAdd(&stats[64 + fg + 2], q2); atomicAdd(&stats[64 + fg + 3], q3);
}
__global__ void bn_apply_k(const float* __restrict__ agg, const float* __restrict__ stats,
                           float* __restrict__ hout, int n, int do_relu) {
    int idx = blockIdx.x * blockDim.x + threadIdx.x;
    if (idx < n * 16) {
        int i = idx >> 4, fg = (idx & 15) * 4;
        float invn = 1.0f / (float)n;
        float4 sm = *(const float4*)(stats + fg);
        float4 sq = *(const float4*)(stats + 64 + fg);
        float m0 = sm.x * invn, m1 = sm.y * invn, m2 = sm.z * invn, m3 = sm.w * invn;
        float r0 = rsqrtf(sq.x * invn - m0 * m0 + BN_EPS);
        float r1 = rsqrtf(sq.y * invn - m1 * m1 + BN_EPS);
        float r2 = rsqrtf(sq.z * invn - m2 * m2 + BN_EPS);
        float r3 = rsqrtf(sq.w * invn - m3 * m3 + BN_EPS);
        float4 v = *(const float4*)(agg + i * 64 + fg);
        if (do_relu) {
            v.x = fmaxf(v.x, 0.f); v.y = fmaxf(v.y, 0.f);
            v.z = fmaxf(v.z, 0.f); v.w = fmaxf(v.w, 0.f);
        }
        float4 o = make_float4((v.x - m0) * r0, (v.y - m1) * r1,
                               (v.z - m2) * r2, (v.w - m3) * r3);
        *(float4*)(hout + i * 64 + fg) = o;
    }
}

// ---------------- global mean pool + classifier + softmax ----------------
__global__ void pool_accum_k(const float* __restrict__ h, const int* __restrict__ batch,
                             float* __restrict__ pooled, float* __restrict__ counts, int n) {
    int idx = blockIdx.x * blockDim.x + threadIdx.x;
    if (idx < n * 16) {
        int i = idx >> 4, fg = (idx & 15) * 4;
        int g = batch[i];
        float4 v = *(const float4*)(h + i * 64 + fg);
        float* p = pooled + g * 64 + fg;
        atomicAdd(p + 0, v.x); atomicAdd(p + 1, v.y);
        atomicAdd(p + 2, v.z); atomicAdd(p + 3, v.w);
        if (fg == 0) atomicAdd(&counts[g], 1.0f);
    }
}
__global__ void classify_k(const float* __restrict__ pooled, const float* __restrict__ counts,
                           const float* __restrict__ Wc, const float* __restrict__ bc,
                           float* __restrict__ out) {
    int g = threadIdx.x;                 // one graph per lane, 64 graphs
    if (g >= N_GRAPHS) return;
    float inv = 1.0f / fmaxf(counts[g], 1.0f);
    float logits[N_CLASSES];
    #pragma unroll
    for (int c = 0; c < N_CLASSES; ++c) logits[c] = bc[c];
    for (int f = 0; f < 64; ++f) {
        float p = pooled[g * 64 + f] * inv;
        #pragma unroll
        for (int c = 0; c < N_CLASSES; ++c) logits[c] += p * Wc[f * N_CLASSES + c];
    }
    float mx = logits[0];
    #pragma unroll
    for (int c = 1; c < N_CLASSES; ++c) mx = fmaxf(mx, logits[c]);
    float sum = 0.0f;
    #pragma unroll
    for (int c = 0; c < N_CLASSES; ++c) { logits[c] = expf(logits[c] - mx); sum += logits[c]; }
    float isum = 1.0f / sum;
    #pragma unroll
    for (int c = 0; c < N_CLASSES; ++c) out[g * N_CLASSES + c] = logits[c] * isum;
}

extern "C" void kernel_launch(void* const* d_in, const int* in_sizes, int n_in,
                              void* d_out, int out_size, void* d_ws, size_t ws_size,
                              hipStream_t stream) {
    const float* x  = (const float*)d_in[0];
    const float* W0 = (const float*)d_in[1];
    const float* b0 = (const float*)d_in[2];
    const float* W1 = (const float*)d_in[3];
    const float* b1 = (const float*)d_in[4];
    const float* W2 = (const float*)d_in[5];
    const float* b2 = (const float*)d_in[6];
    const float* Wc = (const float*)d_in[7];
    const float* bc = (const float*)d_in[8];
    const int* edge_index = (const int*)d_in[9];
    const int* batch      = (const int*)d_in[10];

    const int n = in_sizes[0] / N_FEAT;
    const int E = in_sizes[9] / 2;
    const int* esrc = edge_index;
    const int* edst = edge_index + E;

    // workspace carve-out
    char* w = (char*)d_ws;
    size_t off = 0;
    auto carve = [&](size_t bytes) -> float* {
        float* p = (float*)(w + off);
        off = (off + bytes + 255) & ~(size_t)255;
        return p;
    };
    float* deg    = carve((size_t)n * 4);
    float* isq    = carve((size_t)n * 4);
    float* nrm    = carve((size_t)E * 4);
    float* hl     = carve((size_t)n * 64 * 4);
    float* agg    = carve((size_t)n * 64 * 4);
    float* hcur   = carve((size_t)n * 64 * 4);
    float* stats  = carve(128 * 4);
    float* pooled = carve(N_GRAPHS * 64 * 4);
    float* counts = carve(N_GRAPHS * 4);
    (void)ws_size;

    const int TB = 256;
    const int nv = n * 16;   // node-feature float4 work items
    const int ev = E * 16;   // edge-feature float4 work items

    deg_init_k <<<(n + TB - 1) / TB, TB, 0, stream>>>(deg, n);
    deg_accum_k<<<(E + TB - 1) / TB, TB, 0, stream>>>(edst, deg, E);
    isq_k      <<<(n + TB - 1) / TB, TB, 0, stream>>>(deg, isq, n);
    norm_k     <<<(E + TB - 1) / TB, TB, 0, stream>>>(esrc, edst, isq, nrm, E);

    const float* hin = x;
    for (int layer = 0; layer < 3; ++layer) {
        const float* Wl = (layer == 0) ? W0 : (layer == 1) ? W1 : W2;
        const float* bl = (layer == 0) ? b0 : (layer == 1) ? b1 : b2;
        const int do_relu = (layer < 2) ? 1 : 0;

        gemm_wmma_k     <<<(n + TILE_M - 1) / TILE_M, TB, 0, stream>>>(hin, Wl, hl, n);
        selfloop_bias_k <<<(nv + TB - 1) / TB, TB, 0, stream>>>(hl, isq, bl, agg, n);
        scatter_k       <<<(ev + TB - 1) / TB, TB, 0, stream>>>(esrc, edst, nrm, hl, agg, ev);
        zero_k          <<<1, 128, 0, stream>>>(stats, 128);
        bn_stats_k      <<<512, TB, 0, stream>>>(agg, stats, n, do_relu);
        bn_apply_k      <<<(nv + TB - 1) / TB, TB, 0, stream>>>(agg, stats, hcur, n, do_relu);
        hin = hcur;
    }

    zero_k       <<<(N_GRAPHS * 64 + TB - 1) / TB, TB, 0, stream>>>(pooled, N_GRAPHS * 64);
    zero_k       <<<1, N_GRAPHS, 0, stream>>>(counts, N_GRAPHS);
    pool_accum_k <<<(nv + TB - 1) / TB, TB, 0, stream>>>(hcur, batch, pooled, counts, n);
    classify_k   <<<1, 64, 0, stream>>>(pooled, counts, Wc, bc, (float*)d_out);
}
